// LCGNConceptVocabulary_7344394076518
// MI455X (gfx1250) — matrix operand order repaired and checked by reading, not compiled, not run on hardware
//
#include <hip/hip_runtime.h>
#include <cstdint>

// ---------------------------------------------------------------------------
// Types for CDNA5 WMMA (wave32): v_wmma_f32_16x16x32_bf16
// ---------------------------------------------------------------------------
typedef __bf16 bhalf;
typedef __attribute__((ext_vector_type(16))) __bf16 v16bf;
typedef __attribute__((ext_vector_type(8)))  __bf16 v8bf;
typedef __attribute__((ext_vector_type(8)))  float  v8f;

// CDNA5 async global->LDS copy (ASYNCcnt-tracked, no VGPR round trip).
// LDS byte address comes from the vdst VGPR; 16 bytes per lane.
__device__ __forceinline__ void async_ld16(uint32_t lds_off, const bhalf* gptr) {
  asm volatile("global_load_async_to_lds_b128 %0, %1, off"
               :: "v"(lds_off), "v"((uint64_t)(uintptr_t)gptr)
               : "memory");
}

// ---------------------------------------------------------------------------
// Workhorse GEMM:  C[M,N] = epi( A[M,K] @ B^T * alpha + bias )
//   A  : bf16 row-major [M, lda]  (lda%8==0, K-pad zero-filled)
//   Bt : bf16 row-major [N, ldb]  (operand stored N-major: Bt[n][k])
//   Kp : padded K (multiple of 32, pads zero)
//   FULL: host guarantees M%128==0, N%64==0, N==ldc -> branch-free async path
//   epilogue: *alpha, +bias[n], optional ELU, optional row-vector multiply
//   OUT_TR: C[(gr/trRows)*trBStride + gc*ldc + gr%trRows]  (vals^T)
//   normal: C[gr*ldc+gc]; cols in [N,ldc) zero-filled
//
// Block = 256 threads = 8 wave32.  Tile 128x64x32, LDS double-buffered.
// Interior blocks stream the next K-slice with global_load_async_to_lds_b128
// (ping-pong, s_wait_asynccnt) while WMMAs consume the current slice.
// ---------------------------------------------------------------------------
template <bool FULL, bool ELU, bool OUT_TR, typename OutT>
__global__ void __launch_bounds__(256)
k_gemm(const bhalf* __restrict__ A, int lda,
       const bhalf* __restrict__ Bt, int ldb,
       OutT* __restrict__ C, int ldc,
       const float* __restrict__ bias,
       const float* __restrict__ mul, int mulDiv, int mulStride,
       int M, int N, int Kp, float alpha,
       long long sA, long long sB, long long sC,
       int trRows, long long trBStride)
{
  __shared__ alignas(16) bhalf As[2][128 * 40];  // [m][k], stride 40
  __shared__ alignas(16) bhalf Bs[2][64 * 40];   // [n][k]

  const int tid  = threadIdx.x;
  const int lane = tid & 31;
  const int wv   = tid >> 5;
  const int m0   = blockIdx.y * 128;
  const int n0   = blockIdx.x * 64;

  A  += (size_t)blockIdx.z * sA;
  Bt += (size_t)blockIdx.z * sB;
  C  += (size_t)blockIdx.z * sC;

  const int half = lane >> 4;
  const int l16  = lane & 15;
  const bool fullTile = FULL ? true : ((m0 + 128 <= M) && (n0 + 64 <= N));

  v8f acc[4];
#pragma unroll
  for (int j = 0; j < 4; ++j)
#pragma unroll
    for (int i = 0; i < 8; ++i) acc[j][i] = 0.f;

  v8bf vz;
#pragma unroll
  for (int e = 0; e < 8; ++e) vz[e] = (bhalf)0.f;

  // async staging of one K-slice: 3 async instructions per wave (2xA + 1xB)
  auto stage_async = [&](int buf, int k0) {
#pragma unroll
    for (int c = tid; c < 512; c += 256) {
      int r = c >> 2, k8 = (c & 3) << 3;
      async_ld16((uint32_t)(uintptr_t)&As[buf][r * 40 + k8],
                 &A[(size_t)(m0 + r) * lda + k0 + k8]);
    }
    {
      int n = tid >> 2, k8 = (tid & 3) << 3;
      async_ld16((uint32_t)(uintptr_t)&Bs[buf][n * 40 + k8],
                 &Bt[(size_t)(n0 + n) * ldb + k0 + k8]);
    }
  };
  // guarded synchronous staging for edge tiles (FULL=false only)
  auto stage_sync = [&](int buf, int k0) {
#pragma unroll
    for (int c = tid; c < 512; c += 256) {
      int r = c >> 2, k8 = (c & 3) << 3;
      int gm = m0 + r;
      v8bf v = vz;
      if (gm < M) v = *(const v8bf*)&A[(size_t)gm * lda + k0 + k8];
      *(v8bf*)&As[buf][r * 40 + k8] = v;
    }
    {
      int n = tid >> 2, k8 = (tid & 3) << 3;
      int gn = n0 + n;
      v8bf v = vz;
      if (gn < N) v = *(const v8bf*)&Bt[(size_t)gn * ldb + k0 + k8];
      *(v8bf*)&Bs[buf][n * 40 + k8] = v;
    }
  };

  if (fullTile) stage_async(0, 0); else stage_sync(0, 0);

  int it = 0;
  for (int k0 = 0; k0 < Kp; k0 += 32, ++it) {
    const int cur = it & 1, nxt = cur ^ 1;
    const bool hasNext = (k0 + 32 < Kp);
    if (hasNext) { if (fullTile) stage_async(nxt, k0 + 32); else stage_sync(nxt, k0 + 32); }
    if (fullTile) {
      if (hasNext) asm volatile("s_wait_asynccnt 0x3" ::: "memory");
      else         asm volatile("s_wait_asynccnt 0x0" ::: "memory");
    }
    __syncthreads();

    // A fragment (16x32): row = l16, K sub-ranges by half
    const bhalf* pa = &As[cur][(wv * 16 + l16) * 40 + half * 8];
    v8bf alo = *(const v8bf*)pa;
    v8bf ahi = *(const v8bf*)(pa + 16);
    v16bf afrag;
#pragma unroll
    for (int e = 0; e < 8; ++e) { afrag[e] = alo[e]; afrag[e + 8] = ahi[e]; }

    // 4 B fragments (32x16): one WMMA per 16-col subtile
#pragma unroll
    for (int j = 0; j < 4; ++j) {
      const bhalf* pb = &Bs[cur][(j * 16 + l16) * 40 + half * 8];
      v8bf blo = *(const v8bf*)pb;
      v8bf bhi = *(const v8bf*)(pb + 16);
      v16bf bfrag;
#pragma unroll
      for (int e = 0; e < 8; ++e) { bfrag[e] = blo[e]; bfrag[e + 8] = bhi[e]; }
      acc[j] = __builtin_amdgcn_wmma_f32_16x16x32_bf16(false, afrag, false, bfrag,
                                                       (short)0, acc[j], false, false);
    }
    __syncthreads();
  }

  // epilogue: D VGPR r -> row (half*8 + r), col = l16
#pragma unroll
  for (int j = 0; j < 4; ++j) {
    int gc = n0 + j * 16 + l16;
    const bool cOK = FULL || (gc < N);
    float bv = (bias && cOK) ? bias[gc] : 0.f;
#pragma unroll
    for (int r = 0; r < 8; ++r) {
      int gr = m0 + wv * 16 + half * 8 + r;
      if (!FULL && gr >= M) continue;
      float val = 0.f;
      if (cOK) {
        val = acc[j][r] * alpha + bv;
        if (ELU) val = val > 0.f ? val : (__expf(val) - 1.f);
        if (mul) val *= mul[(size_t)(gr / mulDiv) * mulStride + gc];
      }
      if (OUT_TR) {
        if (cOK) {
          int b = gr / trRows, rr = gr - b * trRows;
          C[(size_t)b * trBStride + (size_t)gc * ldc + rr] = (OutT)val;
        }
      } else {
        if (FULL) C[(size_t)gr * ldc + gc] = (OutT)val;
        else if (gc < ldc) C[(size_t)gr * ldc + gc] = (OutT)val;
      }
    }
  }
}

// ---------------------------------------------------------------------------
// Helpers
// ---------------------------------------------------------------------------
__global__ void k_cvt(const float* __restrict__ in, bhalf* __restrict__ out, long long n) {
  long long i = (long long)blockIdx.x * blockDim.x + threadIdx.x;
  if (i < n) out[i] = (bhalf)in[i];
}

// fp32 [K,N] row-major -> bf16 [N, ldo] (transposed, K-pad zero-filled)
__global__ void k_cvtT(const float* __restrict__ in, bhalf* __restrict__ out,
                       int K, int N, int ldo) {
  long long i = (long long)blockIdx.x * blockDim.x + threadIdx.x;
  long long total = (long long)N * ldo;
  if (i < total) {
    int n = (int)(i / ldo), k = (int)(i % ldo);
    out[i] = (k < K) ? (bhalf)in[(size_t)k * N + n] : (bhalf)0.f;
  }
}

__global__ void k_zero(bhalf* __restrict__ p, long long n) {
  long long i = (long long)blockIdx.x * blockDim.x + threadIdx.x;
  if (i < n) p[i] = (bhalf)0.f;
}

// bf16 transpose: out[c*ldout + r] = in[r*ldin + c]
__global__ void k_trbf(const bhalf* __restrict__ in, int ldin,
                       bhalf* __restrict__ out, int ldout, int rows, int cols) {
  long long i = (long long)blockIdx.x * blockDim.x + threadIdx.x;
  if (i < (long long)rows * cols) {
    int r = (int)(i / cols), c = (int)(i % cols);
    out[(size_t)c * ldout + r] = in[(size_t)r * ldin + c];
  }
}

// L2-normalize each row, emit bf16 with zero-filled pad columns
__global__ void k_rownorm(const float* __restrict__ in, bhalf* __restrict__ out,
                          int cols, int ldo) {
  const int row = blockIdx.x, tid = threadIdx.x;
  const float* p = in + (size_t)row * cols;
  __shared__ float sh[256];
  float ss = 0.f;
  for (int c = tid; c < cols; c += 256) { float v = p[c]; ss += v * v; }
  sh[tid] = ss; __syncthreads();
  for (int s = 128; s > 0; s >>= 1) { if (tid < s) sh[tid] += sh[tid + s]; __syncthreads(); }
  float inv = 1.f / fmaxf(sqrtf(sh[0]), 1e-12f);
  bhalf* o = out + (size_t)row * ldo;
  for (int c = tid; c < ldo; c += 256)
    o[c] = (c < cols) ? (bhalf)(p[c] * inv) : (bhalf)0.f;
}

__global__ void k_fill_ctx(bhalf* __restrict__ x, const float* __restrict__ initMem,
                           long long total) {
  long long i = (long long)blockIdx.x * blockDim.x + threadIdx.x;
  if (i < total) x[i] = (bhalf)initMem[i & 511];
}

// row softmax fp32 -> bf16 (pad cols zeroed)
__global__ void k_softmax(const float* __restrict__ in, bhalf* __restrict__ out,
                          int cols, int ldo) {
  const int row = blockIdx.x, tid = threadIdx.x;
  const float* p = in + (size_t)row * cols;
  __shared__ float sh[256];
  __shared__ float s_max, s_inv;
  float mx = -3.0e38f;
  for (int c = tid; c < cols; c += 256) mx = fmaxf(mx, p[c]);
  sh[tid] = mx; __syncthreads();
  for (int s = 128; s > 0; s >>= 1) { if (tid < s) sh[tid] = fmaxf(sh[tid], sh[tid + s]); __syncthreads(); }
  if (tid == 0) s_max = sh[0];
  __syncthreads();
  mx = s_max;
  float sum = 0.f;
  for (int c = tid; c < cols; c += 256) sum += __expf(p[c] - mx);
  sh[tid] = sum; __syncthreads();
  for (int s = 128; s > 0; s >>= 1) { if (tid < s) sh[tid] += sh[tid + s]; __syncthreads(); }
  if (tid == 0) s_inv = 1.f / sh[0];
  __syncthreads();
  float inv = s_inv;
  bhalf* o = out + (size_t)row * ldo;
  for (int c = tid; c < ldo; c += 256)
    o[c] = (c < cols) ? (bhalf)(__expf(p[c] - mx) * inv) : (bhalf)0.f;
}

// masked edge softmax: rows = B*196, valid cols m < entity_num[b]; pads zeroed
__global__ void k_softmax_mask(const float* __restrict__ in, bhalf* __restrict__ out,
                               const int* __restrict__ ent, int cols, int ldo) {
  const int row = blockIdx.x, tid = threadIdx.x;
  const int b = row / 196;
  const int nv = ent[b];
  const float* p = in + (size_t)row * cols;
  __shared__ float sh[256];
  __shared__ float s_max, s_inv;
  float mx = -3.0e38f;
  for (int c = tid; c < cols; c += 256) if (c < nv) mx = fmaxf(mx, p[c]);
  sh[tid] = mx; __syncthreads();
  for (int s = 128; s > 0; s >>= 1) { if (tid < s) sh[tid] = fmaxf(sh[tid], sh[tid + s]); __syncthreads(); }
  if (tid == 0) s_max = sh[0];
  __syncthreads();
  mx = s_max;
  float sum = 0.f;
  for (int c = tid; c < cols; c += 256) if (c < nv) sum += __expf(p[c] - mx);
  sh[tid] = sum; __syncthreads();
  for (int s = 128; s > 0; s >>= 1) { if (tid < s) sh[tid] += sh[tid + s]; __syncthreads(); }
  if (tid == 0) s_inv = 1.f / sh[0];
  __syncthreads();
  float inv = s_inv;
  bhalf* o = out + (size_t)row * ldo;
  for (int c = tid; c < ldo; c += 256)
    o[c] = (c < nv) ? (bhalf)(__expf(p[c] - mx) * inv) : (bhalf)0.f;
}

// command attention over S=30 -> cmd[b,512] bf16
__global__ void k_cmd_attend(const float* __restrict__ qcmd, const float* __restrict__ lstm,
                             const float* __restrict__ lw, const float* __restrict__ lb,
                             const int* __restrict__ q_len, bhalf* __restrict__ cmd_bf) {
  const int b = blockIdx.x, tid = threadIdx.x;
  __shared__ float att[32];
  __shared__ float s_inv;
  const float* L = lstm + (size_t)b * 30 * 512;
  const float* Q = qcmd + (size_t)b * 512;
  if (tid < 30) {
    float s = 0.f;
    for (int d = 0; d < 512; ++d) s += Q[d] * L[tid * 512 + d] * lw[d];
    att[tid] = s + lb[0];
  }
  __syncthreads();
  if (tid == 0) {
    int ql = q_len[b];
    float mx = -3.0e38f;
    for (int s = 0; s < 30; ++s) if (s < ql) mx = fmaxf(mx, att[s]);
    float sum = 0.f;
    for (int s = 0; s < 30; ++s) {
      float e = (s < ql) ? __expf(att[s] - mx) : 0.f;
      att[s] = e; sum += e;
    }
    s_inv = 1.f / sum;
  }
  __syncthreads();
  float inv = s_inv;
  for (int d = tid; d < 512; d += 256) {
    float s = 0.f;
    for (int t = 0; t < 30; ++t) s += att[t] * L[t * 512 + d];
    cmd_bf[(size_t)b * 512 + d] = (bhalf)(s * inv);
  }
}

// x_joint = [x_loc | x_ctx | proj_loc*proj_ctx]  (bf16, row stride 1536)
__global__ void k_pack_joint(const bhalf* __restrict__ xl, const bhalf* __restrict__ xc,
                             const float* __restrict__ pl, const float* __restrict__ pc,
                             bhalf* __restrict__ xj, long long total) {
  long long i = (long long)blockIdx.x * blockDim.x + threadIdx.x;
  if (i < total) {
    long long r = i >> 9; int d = (int)(i & 511);
    bhalf* o = xj + r * 1536;
    o[d]        = xl[i];
    o[512 + d]  = xc[i];
    o[1024 + d] = (bhalf)(pl[i] * pc[i]);
  }
}

// cat = [a | b] (bf16, row stride 1024)
__global__ void k_pack2(const bhalf* __restrict__ a, const bhalf* __restrict__ b,
                        bhalf* __restrict__ dst, long long total) {
  long long i = (long long)blockIdx.x * blockDim.x + threadIdx.x;
  if (i < total) {
    long long r = i >> 9; int d = (int)(i & 511);
    bhalf* o = dst + r * 1024;
    o[d]       = a[i];
    o[512 + d] = b[i];
  }
}

// ---------------------------------------------------------------------------
// Host orchestration
// ---------------------------------------------------------------------------
extern "C" void kernel_launch(void* const* d_in, const int* in_sizes, int n_in,
                              void* d_out, int out_size, void* d_ws, size_t ws_size,
                              hipStream_t stream) {
  (void)in_sizes; (void)n_in; (void)out_size; (void)ws_size;

  constexpr int B = 64, N = 196, DF = 2112, MD = 300, CD = 2000, CTX = 512, T = 4;
  constexpr int R = B * N;                 // 12544 = 98*128
  constexpr int MDp = 320;                 // 300 -> pad 32
  constexpr int CDp = 2016;                // 2000 -> pad 32
  constexpr int Np  = 224;                 // 196 -> pad 32
  const float inv_sqrt_m   = 0.057735026919f;   // 1/sqrt(300)
  const float inv_sqrt_ctx = 0.044194173824f;   // 1/sqrt(512)

  const float* images = (const float*)d_in[0];
  const float* q_enc  = (const float*)d_in[1];
  const float* lstm   = (const float*)d_in[2];
  const float* emb    = (const float*)d_in[3];
  const int*   q_len  = (const int*)d_in[4];
  const int*   ent    = (const int*)d_in[5];
  const float* cpt_attn_w = (const float*)d_in[7];
  const float* cpt_attn_b = (const float*)d_in[8];
  const float* initCKB_w  = (const float*)d_in[9];
  const float* initCKB_b  = (const float*)d_in[10];
  const float* initMem    = (const float*)d_in[11];
  const float* qInput_w   = (const float*)d_in[12];
  const float* qInput_b   = (const float*)d_in[13];
  const float* qInput2_w  = (const float*)d_in[14];
  const float* qInput2_b  = (const float*)d_in[15];
  const float* logits_w   = (const float*)d_in[16];
  const float* logits_b   = (const float*)d_in[17];
  const float* ploc_w = (const float*)d_in[18];
  const float* ploc_b = (const float*)d_in[19];
  const float* pctx_w = (const float*)d_in[20];
  const float* pctx_b = (const float*)d_in[21];
  const float* q_w = (const float*)d_in[22];
  const float* q_b = (const float*)d_in[23];
  const float* kw_ = (const float*)d_in[24];
  const float* k_b = (const float*)d_in[25];
  const float* v_w = (const float*)d_in[26];
  const float* v_b = (const float*)d_in[27];
  const float* pk_w = (const float*)d_in[28];
  const float* pk_b = (const float*)d_in[29];
  const float* pv_w = (const float*)d_in[30];
  const float* pv_b = (const float*)d_in[31];
  const float* mem_w = (const float*)d_in[32];
  const float* mem_b = (const float*)d_in[33];
  const float* comb_w = (const float*)d_in[34];
  const float* comb_b = (const float*)d_in[35];

  char* base = (char*)d_ws;
  size_t off = 0;
  auto alloc = [&](size_t bytes) -> void* {
    off = (off + 255) & ~(size_t)255;
    void* p = base + off;
    off += bytes;
    return p;
  };

  // persistent: transposed bf16 weights ([N, ldK] layout)
  bhalf* w_cptT  = (bhalf*)alloc((size_t)MD * DF * 2);          // [300,2112]
  bhalf* w_ckbT  = (bhalf*)alloc((size_t)CTX * MDp * 2);        // [512,320]
  bhalf* w_qiT   = (bhalf*)alloc((size_t)CTX * CTX * 2);
  bhalf* w_qi2T  = (bhalf*)alloc((size_t)T * CTX * CTX * 2);
  bhalf* w_plocT = (bhalf*)alloc((size_t)CTX * CTX * 2);
  bhalf* w_pctxT = (bhalf*)alloc((size_t)CTX * CTX * 2);
  bhalf* w_qT    = (bhalf*)alloc((size_t)CTX * 3 * CTX * 2);    // [512,1536]
  bhalf* w_kT    = (bhalf*)alloc((size_t)CTX * 3 * CTX * 2);
  bhalf* w_vT    = (bhalf*)alloc((size_t)CTX * 3 * CTX * 2);
  bhalf* w_pkT   = (bhalf*)alloc((size_t)CTX * CTX * 2);
  bhalf* w_pvT   = (bhalf*)alloc((size_t)CTX * CTX * 2);
  bhalf* w_memT  = (bhalf*)alloc((size_t)CTX * 2 * CTX * 2);    // [512,1024]
  bhalf* w_combT = (bhalf*)alloc((size_t)CTX * 2 * CTX * 2);
  // persistent activations
  bhalf* embn    = (bhalf*)alloc((size_t)CD * MDp * 2);         // [2000,320]
  bhalf* embnT   = (bhalf*)alloc((size_t)MD * CDp * 2);         // [300,2016]
  bhalf* xloc    = (bhalf*)alloc((size_t)R * CTX * 2);
  bhalf* xctx    = (bhalf*)alloc((size_t)R * CTX * 2);
  bhalf* qenc_bf = (bhalf*)alloc((size_t)B * CTX * 2);
  bhalf* h_bf    = (bhalf*)alloc((size_t)B * CTX * 2);
  bhalf* cmd_bf  = (bhalf*)alloc((size_t)B * CTX * 2);
  float* qcmd_f  = (float*)alloc((size_t)B * CTX * 4);
  float* pk_f    = (float*)alloc((size_t)B * CTX * 4);
  float* pv_f    = (float*)alloc((size_t)B * CTX * 4);

  // phase-1 arena (concept vocabulary)
  size_t mark = off;
  bhalf* imgs_bf = (bhalf*)alloc((size_t)R * DF * 2);
  bhalf* imgq_bf = (bhalf*)alloc((size_t)R * MDp * 2);          // [R,320]
  float* score_f = (float*)alloc((size_t)R * CD * 4);           // [R,2000]
  bhalf* prob_bf = (bhalf*)alloc((size_t)R * CDp * 2);          // [R,2016]
  bhalf* xlocm   = (bhalf*)alloc((size_t)R * MDp * 2);          // [R,320]
  // phase-2 arena aliases phase-1
  off = mark;
  float* ploc_f  = (float*)alloc((size_t)R * CTX * 4);
  float* pctx_f  = (float*)alloc((size_t)R * CTX * 4);
  bhalf* xj_bf   = (bhalf*)alloc((size_t)R * 3 * CTX * 2);
  bhalf* qrs_bf  = (bhalf*)alloc((size_t)R * CTX * 2);
  bhalf* keys_bf = (bhalf*)alloc((size_t)R * CTX * 2);
  bhalf* valsT   = (bhalf*)alloc((size_t)B * CTX * Np * 2);     // [B,512,224]
  float* esc_f   = (float*)alloc((size_t)B * N * N * 4);
  bhalf* eprob   = (bhalf*)alloc((size_t)R * Np * 2);           // [B*196,224]
  bhalf* msg_bf  = (bhalf*)alloc((size_t)R * CTX * 2);
  bhalf* cat_bf  = (bhalf*)alloc((size_t)R * 2 * CTX * 2);

  const dim3 blk(256);
  auto g1 = [](long long n) { return dim3((unsigned)((n + 255) / 256)); };
  auto cvtT = [&](const float* s, bhalf* d, int K, int Nc, int ldo) {
    k_cvtT<<<g1((long long)Nc * ldo), blk, 0, stream>>>(s, d, K, Nc, ldo);
  };
  auto gg = [](int Mr, int NcCover, int bat) {
    return dim3((unsigned)((NcCover + 63) / 64), (unsigned)((Mr + 127) / 128), (unsigned)bat);
  };

  // ---- weight conversion (transposed, K-padded) ----
  cvtT(cpt_attn_w, w_cptT, DF, MD, DF);
  cvtT(initCKB_w,  w_ckbT, MD, CTX, MDp);
  cvtT(qInput_w,   w_qiT,  CTX, CTX, CTX);
  for (int t = 0; t < T; ++t)
    cvtT(qInput2_w + (size_t)t * CTX * CTX, w_qi2T + (size_t)t * CTX * CTX, CTX, CTX, CTX);
  cvtT(ploc_w, w_plocT, CTX, CTX, CTX);
  cvtT(pctx_w, w_pctxT, CTX, CTX, CTX);
  cvtT(q_w, w_qT, 3 * CTX, CTX, 3 * CTX);
  cvtT(kw_, w_kT, 3 * CTX, CTX, 3 * CTX);
  cvtT(v_w, w_vT, 3 * CTX, CTX, 3 * CTX);
  cvtT(pk_w, w_pkT, CTX, CTX, CTX);
  cvtT(pv_w, w_pvT, CTX, CTX, CTX);
  cvtT(mem_w,  w_memT,  2 * CTX, CTX, 2 * CTX);
  cvtT(comb_w, w_combT, 2 * CTX, CTX, 2 * CTX);
  k_cvt<<<g1(B * CTX), blk, 0, stream>>>(q_enc, qenc_bf, (long long)B * CTX);

  // pad hygiene (replays start from poisoned ws)
  k_zero<<<g1((long long)MD * CDp), blk, 0, stream>>>(embnT, (long long)MD * CDp);
  k_zero<<<g1((long long)B * CTX * Np), blk, 0, stream>>>(valsT, (long long)B * CTX * Np);

  // ---- loc_ctx_init ----
  k_rownorm<<<R,  blk, 0, stream>>>(images, imgs_bf, DF, DF);
  k_rownorm<<<CD, blk, 0, stream>>>(emb, embn, MD, MDp);
  k_trbf<<<g1((long long)CD * MD), blk, 0, stream>>>(embn, MDp, embnT, CDp, CD, MD);

  // img_queries = imgs @ cpt_attn_w + b              [R,300] (ld 320)
  k_gemm<false, false, false, bhalf><<<gg(R, MDp, 1), blk, 0, stream>>>(
      imgs_bf, DF, w_cptT, DF, imgq_bf, MDp, cpt_attn_b, nullptr, 1, 1,
      R, MD, DF, 1.f, 0, 0, 0, 1, 0);
  // cpt_score = img_queries @ embn^T / sqrt(300)     [R,2000]
  k_gemm<false, false, false, float><<<gg(R, CD, 1), blk, 0, stream>>>(
      imgq_bf, MDp, embn, MDp, score_f, CD, nullptr, nullptr, 1, 1,
      R, CD, MDp, inv_sqrt_m, 0, 0, 0, 1, 0);
  k_softmax<<<R, blk, 0, stream>>>(score_f, prob_bf, CD, CDp);
  // x_loc_m = cpt_prob @ embn                        [R,300] (ld 320)
  k_gemm<false, false, false, bhalf><<<gg(R, MDp, 1), blk, 0, stream>>>(
      prob_bf, CDp, embnT, CDp, xlocm, MDp, nullptr, nullptr, 1, 1,
      R, MD, CDp, 1.f, 0, 0, 0, 1, 0);
  // x_loc = x_loc_m @ initCKB_w + b                  [R,512]  (FULL)
  k_gemm<true, false, false, bhalf><<<gg(R, CTX, 1), blk, 0, stream>>>(
      xlocm, MDp, w_ckbT, MDp, xloc, CTX, initCKB_b, nullptr, 1, 1,
      R, CTX, MDp, 1.f, 0, 0, 0, 1, 0);
  k_fill_ctx<<<g1((long long)R * CTX), blk, 0, stream>>>(xctx, initMem, (long long)R * CTX);

  // ---- T message-passing iterations ----
  for (int t = 0; t < T; ++t) {
    k_gemm<false, true, false, bhalf><<<gg(B, CTX, 1), blk, 0, stream>>>(
        qenc_bf, CTX, w_qiT, CTX, h_bf, CTX, qInput_b, nullptr, 1, 1,
        B, CTX, CTX, 1.f, 0, 0, 0, 1, 0);
    k_gemm<false, false, false, float><<<gg(B, CTX, 1), blk, 0, stream>>>(
        h_bf, CTX, w_qi2T + (size_t)t * CTX * CTX, CTX, qcmd_f, CTX,
        qInput2_b + (size_t)t * CTX, nullptr, 1, 1, B, CTX, CTX, 1.f, 0, 0, 0, 1, 0);
    k_cmd_attend<<<B, blk, 0, stream>>>(qcmd_f, lstm, logits_w, logits_b, q_len, cmd_bf);
    k_gemm<false, false, false, float><<<gg(B, CTX, 1), blk, 0, stream>>>(
        cmd_bf, CTX, w_pkT, CTX, pk_f, CTX, pk_b, nullptr, 1, 1,
        B, CTX, CTX, 1.f, 0, 0, 0, 1, 0);
    k_gemm<false, false, false, float><<<gg(B, CTX, 1), blk, 0, stream>>>(
        cmd_bf, CTX, w_pvT, CTX, pv_f, CTX, pv_b, nullptr, 1, 1,
        B, CTX, CTX, 1.f, 0, 0, 0, 1, 0);

    k_gemm<true, false, false, float><<<gg(R, CTX, 1), blk, 0, stream>>>(
        xloc, CTX, w_plocT, CTX, ploc_f, CTX, ploc_b, nullptr, 1, 1,
        R, CTX, CTX, 1.f, 0, 0, 0, 1, 0);
    k_gemm<true, false, false, float><<<gg(R, CTX, 1), blk, 0, stream>>>(
        xctx, CTX, w_pctxT, CTX, pctx_f, CTX, pctx_b, nullptr, 1, 1,
        R, CTX, CTX, 1.f, 0, 0, 0, 1, 0);
    k_pack_joint<<<g1((long long)R * CTX), blk, 0, stream>>>(
        xloc, xctx, ploc_f, pctx_f, xj_bf, (long long)R * CTX);

    k_gemm<true, false, false, bhalf><<<gg(R, CTX, 1), blk, 0, stream>>>(
        xj_bf, 3 * CTX, w_qT, 3 * CTX, qrs_bf, CTX, q_b, nullptr, 1, 1,
        R, CTX, 3 * CTX, 1.f, 0, 0, 0, 1, 0);
    k_gemm<true, false, false, bhalf><<<gg(R, CTX, 1), blk, 0, stream>>>(
        xj_bf, 3 * CTX, w_kT, 3 * CTX, keys_bf, CTX, k_b, pk_f, N, CTX,
        R, CTX, 3 * CTX, 1.f, 0, 0, 0, 1, 0);
    // vals written transposed per batch: valsT[b][d][n], ld 224
    k_gemm<true, false, true, bhalf><<<gg(R, CTX, 1), blk, 0, stream>>>(
        xj_bf, 3 * CTX, w_vT, 3 * CTX, valsT, Np, v_b, pv_f, N, CTX,
        R, CTX, 3 * CTX, 1.f, 0, 0, 0, N, (long long)CTX * Np);

    // edge_score[b] = Q_b @ K_b^T / sqrt(512)
    k_gemm<false, false, false, float><<<gg(N, N, B), blk, 0, stream>>>(
        qrs_bf, CTX, keys_bf, CTX, esc_f, N, nullptr, nullptr, 1, 1,
        N, N, CTX, inv_sqrt_ctx,
        (long long)N * CTX, (long long)N * CTX, (long long)N * N, 1, 0);
    k_softmax_mask<<<R, blk, 0, stream>>>(esc_f, eprob, ent, N, Np);
    // message[b] = P_b @ V_b   (B operand = valsT, already [512,224])
    k_gemm<false, false, false, bhalf><<<gg(N, CTX, B), blk, 0, stream>>>(
        eprob, Np, valsT, Np, msg_bf, CTX, nullptr, nullptr, 1, 1,
        N, CTX, Np, 1.f,
        (long long)N * Np, (long long)CTX * Np, (long long)N * CTX, 1, 0);

    k_pack2<<<g1((long long)R * CTX), blk, 0, stream>>>(
        xctx, msg_bf, cat_bf, (long long)R * CTX);
    k_gemm<true, false, false, bhalf><<<gg(R, CTX, 1), blk, 0, stream>>>(
        cat_bf, 2 * CTX, w_memT, 2 * CTX, xctx, CTX, mem_b, nullptr, 1, 1,
        R, CTX, 2 * CTX, 1.f, 0, 0, 0, 1, 0);
  }

  // ---- x_out = [x_loc | x_ctx] @ comb_w + b -> d_out (fp32) ----
  k_pack2<<<g1((long long)R * CTX), blk, 0, stream>>>(
      xloc, xctx, cat_bf, (long long)R * CTX);
  k_gemm<true, false, false, float><<<gg(R, CTX, 1), blk, 0, stream>>>(
      cat_bf, 2 * CTX, w_combT, 2 * CTX, (float*)d_out, CTX, comb_b, nullptr, 1, 1,
      R, CTX, 2 * CTX, 1.f, 0, 0, 0, 1, 0);
}
